// LightplaneRenderer_31353261261022
// MI455X (gfx1250) — matrix operand order repaired and checked by reading, not compiled, not run on hardware
//
#include <hip/hip_runtime.h>
#include <hip/hip_bf16.h>
#include <math.h>

typedef __attribute__((ext_vector_type(16))) _Float16 v16h;
typedef __attribute__((ext_vector_type(8)))  _Float16 v8h;
typedef __attribute__((ext_vector_type(4)))  _Float16 v4h;
typedef __attribute__((ext_vector_type(8)))  float    v8f;

#define RR   8192
#define SSN  128
#define CCH  32
#define HIDN 64
#define GGR  128
#define GRID_ELEMS (GGR * GGR * GGR * CCH)          // 67,108,864
#define GRID_F16_BYTES ((size_t)GRID_ELEMS * 2)     // 128 MiB: fits 192 MB L2

// f16 weight blob in LDS, pre-permuted into WMMA B-operand layout
#define OFF_WT1 0
#define OFF_WT2 2048
#define OFF_WO1 6144
#define OFF_WC1 10240
#define WB_TOT  14336

__device__ __forceinline__ int clampi(int v, int lo, int hi) {
  return v < lo ? lo : (v > hi ? hi : v);
}

// B operand (32x16 f16 tile): lane<16 holds K={0..7,16..23} of column N=lane, lane>=16 holds
// K={8..15,24..31} of column N=lane-16; pre-permuted so each lane reads 16 contiguous f16 (32B).
__device__ __forceinline__ v16h load_b(const _Float16* wB, int base, int kt, int nt, int lane) {
  return *(const v16h*)(wB + base + (((kt * 4 + nt) * 32 + lane) << 4));
}

// A operand (16x32 f16 tile) from row-major [m][64] f16 activations in LDS.
// lane<16: M=lane, K={0..7,16..23}; lane>=16: M=lane-16, K={8..15,24..31} (+kofs).
__device__ __forceinline__ v16h load_a(const _Float16* act, int mrow, int kofs, int kbA) {
  v8h lo = *(const v8h*)(act + mrow * 64 + kofs + kbA);
  v8h hi = *(const v8h*)(act + mrow * 64 + kofs + kbA + 16);
  v16h a;
#pragma unroll
  for (int i = 0; i < 8; ++i) { a[i] = lo[i]; a[i + 8] = hi[i]; }
  return a;
}

// One 64->64 layer: D = relu(act(16x64) @ W(64x64) + b). K=64 split into two
// v_wmma_f32_16x16x32_f16 per 16-wide N tile. Result written back as f16 (next
// WMMA A operand) or as f32 scratch (for the scalar heads).
__device__ __forceinline__ void layer64(_Float16* act, float* fbuf, const _Float16* wB,
                                        int wbase, const float* __restrict__ biasG,
                                        int lane, int mrow, int kbA, bool toF16) {
  v16h aA = load_a(act, mrow, 0, kbA);
  v16h aB = load_a(act, mrow, 32, kbA);
  v8f acc[4];
#pragma unroll
  for (int nt = 0; nt < 4; ++nt) {
    v8f c = {};
    c = __builtin_amdgcn_wmma_f32_16x16x32_f16(false, aA, false,
            load_b(wB, wbase, 0, nt, lane), (short)0, c, false, false);
    c = __builtin_amdgcn_wmma_f32_16x16x32_f16(false, aB, false,
            load_b(wB, wbase, 1, nt, lane), (short)0, c, false, false);
    acc[nt] = c;
  }
#pragma unroll
  for (int nt = 0; nt < 4; ++nt) {
    int n = nt * 16 + (lane & 15);
    float bn = biasG[n];
#pragma unroll
    for (int r = 0; r < 8; ++r) {
      int m = r + ((lane >> 4) << 3);   // C-layout: lanes 0-15 -> M=r, lanes 16-31 -> M=8+r
      float v = fmaxf(acc[nt][r] + bn, 0.f);
      if (toF16) act[m * 64 + n] = (_Float16)v;
      else       fbuf[m * 64 + n] = v;
    }
  }
}

// ---- corner loaders: 16 channels (this lane's A-fragment channel set) ----
__device__ __forceinline__ void load_corner(const float* basep, float q[16]) {
  const float4* b4 = (const float4*)basep;
  float4 q0 = b4[0], q1 = b4[1], q2 = b4[4], q3 = b4[5];
  q[0]=q0.x; q[1]=q0.y; q[2]=q0.z; q[3]=q0.w;
  q[4]=q1.x; q[5]=q1.y; q[6]=q1.z; q[7]=q1.w;
  q[8]=q2.x; q[9]=q2.y; q[10]=q2.z; q[11]=q2.w;
  q[12]=q3.x; q[13]=q3.y; q[14]=q3.z; q[15]=q3.w;
}
__device__ __forceinline__ void load_corner(const _Float16* basep, float q[16]) {
  v8h h0 = *(const v8h*)basep;          // ch kbA+0..7   (16B)
  v8h h1 = *(const v8h*)(basep + 16);   // ch kbA+16..23 (16B)
#pragma unroll
  for (int i = 0; i < 8; ++i) { q[i] = (float)h0[i]; q[8 + i] = (float)h1[i]; }
}

// ---- pre-pass: feature grid f32 -> f16 (halves it to 128 MB => L2-resident) ----
__global__ void __launch_bounds__(256)
grid_to_f16(const float* __restrict__ src, _Float16* __restrict__ dst, int n4) {
  int stride = gridDim.x * blockDim.x;
  const float4* s4 = (const float4*)src;
  v4h* d4 = (v4h*)dst;
  for (int i = blockIdx.x * blockDim.x + threadIdx.x; i < n4; i += stride) {
    float4 v = s4[i];
    v4h o;
    o[0] = (_Float16)v.x; o[1] = (_Float16)v.y;
    o[2] = (_Float16)v.z; o[3] = (_Float16)v.w;
    d4[i] = o;
  }
}

template <typename GT>
__global__ void __launch_bounds__(128)
lightplane_render(const float* __restrict__ org, const float* __restrict__ dir,
                  const float* __restrict__ nearG, const float* __restrict__ farG,
                  const GT* __restrict__ grid,
                  const float* __restrict__ WrG, const float* __restrict__ brG,
                  const float* __restrict__ Wt1G, const float* __restrict__ bt1G,
                  const float* __restrict__ Wt2G, const float* __restrict__ bt2G,
                  const float* __restrict__ Wo1G, const float* __restrict__ bo1G,
                  const float* __restrict__ Wo2G, const float* __restrict__ bo2G,
                  const float* __restrict__ Wc1G, const float* __restrict__ bc1G,
                  const float* __restrict__ Wc2G, const float* __restrict__ bc2G,
                  float* __restrict__ out)
{
  __shared__ __align__(32) _Float16 wB[WB_TOT];
  __shared__ __align__(32) _Float16 actAll[4][16 * 64];
  __shared__ float fbufAll[4][16 * 64];
  __shared__ float alphaS[SSN];
  __shared__ float colS[SSN * 3];
  __shared__ float rayenc[HIDN];

  const int ray  = blockIdx.x;
  const int tid  = threadIdx.x;
  const int lane = tid & 31;
  const int wv   = tid >> 5;
  const int mrow = lane & 15;
  const int kbA  = (lane < 16) ? 0 : 8;

  // ---- stage f32 weights -> f16 LDS in WMMA B-operand layout ----
  for (int idx = tid; idx < WB_TOT; idx += blockDim.x) {
    const float* W; int local, kt;
    if (idx < OFF_WT2) { W = Wt1G; local = idx; kt = 0; }
    else {
      int r = idx - OFF_WT2;
      int m = r >> 12;                 // 0:Wt2 1:Wo1 2:Wc1
      local = r & 4095;
      kt = (local >> 11) & 1;
      W = (m == 0) ? Wt2G : (m == 1) ? Wo1G : Wc1G;
    }
    int t  = local & 1;
    int j  = (local >> 1) & 7;
    int ln = (local >> 4) & 31;
    int nt = (local >> 9) & 3;
    int kb = (ln & 16) ? 8 : 0;
    int k  = kt * 32 + kb + ((j & 4) << 2) + 2 * (j & 3) + t;
    int n  = nt * 16 + (ln & 15);
    wB[idx] = (_Float16)W[k * HIDN + n];
  }

  const float dx = dir[ray * 3 + 0], dy = dir[ray * 3 + 1], dz = dir[ray * 3 + 2];
  const float ox = org[ray * 3 + 0], oy = org[ray * 3 + 1], oz = org[ray * 3 + 2];
  const float nr = nearG[ray], fa = farG[ray];
  const float delta = (fa - nr) / (float)SSN;

  // ---- harmonic ray encoding: rayenc = harmonic(d) @ Wr + br (threads 0..63) ----
  if (tid < HIDN) {
    float s = brG[tid];
    float d3[3] = {dx, dy, dz};
#pragma unroll
    for (int h = 0; h < 3; ++h) {
      float fq = (float)(1 << h);
#pragma unroll
      for (int ax = 0; ax < 3; ++ax) {
        s += sinf(d3[ax] * fq) * WrG[(h * 3 + ax) * HIDN + tid];
        s += cosf(d3[ax] * fq) * WrG[(9 + h * 3 + ax) * HIDN + tid];
      }
    }
#pragma unroll
    for (int ax = 0; ax < 3; ++ax) s += d3[ax] * WrG[(18 + ax) * HIDN + tid];
    rayenc[tid] = s;
  }
  __syncthreads();

  _Float16* act = actAll[wv];
  float*    fbuf = fbufAll[wv];

  for (int rep = 0; rep < 2; ++rep) {
    const int tt = wv * 2 + rep;         // 16-sample tile (8 tiles = 128 samples)
    const int s  = tt * 16 + mrow;       // this lane's sample (M row)

    // ---- trilinear gather: sample s, the 16 channels of this lane's A fragment ----
    float tv = nr + (fa - nr) * ((float)s + 0.5f) / (float)SSN;
    float px = ox + tv * dx, py = oy + tv * dy, pz = oz + tv * dz;
    float cx = (px + 1.f) * 0.5f * (GGR - 1);
    float cy = (py + 1.f) * 0.5f * (GGR - 1);
    float cz = (pz + 1.f) * 0.5f * (GGR - 1);
    float fx = cx - floorf(cx), fy = cy - floorf(cy), fz = cz - floorf(cz);
    int x0 = clampi((int)floorf(cx), 0, GGR - 1);
    int y0 = clampi((int)floorf(cy), 0, GGR - 1);
    int z0 = clampi((int)floorf(cz), 0, GGR - 1);
    int x1 = min(x0 + 1, GGR - 1), y1 = min(y0 + 1, GGR - 1), z1 = min(z0 + 1, GGR - 1);

    // prefetch next tile's corner-0 line (hides latency behind this tile's WMMAs)
    {
      float tv2 = nr + (fa - nr) * ((float)(s + 16) + 0.5f) / (float)SSN;
      float qx = ox + tv2 * dx, qy = oy + tv2 * dy, qz = oz + tv2 * dz;
      int ax = clampi((int)floorf((qx + 1.f) * 0.5f * (GGR - 1)), 0, GGR - 1);
      int ay = clampi((int)floorf((qy + 1.f) * 0.5f * (GGR - 1)), 0, GGR - 1);
      int az = clampi((int)floorf((qz + 1.f) * 0.5f * (GGR - 1)), 0, GGR - 1);
      __builtin_prefetch(grid + ((size_t)((az * GGR + ay) * GGR + ax)) * CCH + kbA, 0, 1);
    }

    float f[16];
#pragma unroll
    for (int i = 0; i < 16; ++i) f[i] = 0.f;
#pragma unroll
    for (int corner = 0; corner < 8; ++corner) {
      int gx = (corner & 1) ? x1 : x0;
      int gy = (corner & 2) ? y1 : y0;
      int gz = (corner & 4) ? z1 : z0;
      float wgt = ((corner & 1) ? fx : 1.f - fx) *
                  ((corner & 2) ? fy : 1.f - fy) *
                  ((corner & 4) ? fz : 1.f - fz);
      const GT* basep = grid + ((size_t)((gz * GGR + gy) * GGR + gx)) * CCH + kbA;
      float q[16];
      load_corner(basep, q);
#pragma unroll
      for (int i = 0; i < 16; ++i) f[i] += wgt * q[i];
    }

    v16h a0;
#pragma unroll
    for (int i = 0; i < 16; ++i) a0[i] = (_Float16)f[i];

    // ---- layer 1: e1 = relu(feats(16x32) @ Wt1(32x64) + bt1) ----
    {
      v8f acc[4];
#pragma unroll
      for (int nt = 0; nt < 4; ++nt) {
        v8f c = {};
        acc[nt] = __builtin_amdgcn_wmma_f32_16x16x32_f16(false, a0, false,
                      load_b(wB, OFF_WT1, 0, nt, lane), (short)0, c, false, false);
      }
#pragma unroll
      for (int nt = 0; nt < 4; ++nt) {
        int n = nt * 16 + (lane & 15);
        float bn = bt1G[n];
#pragma unroll
        for (int r = 0; r < 8; ++r) {
          int m = r + ((lane >> 4) << 3);
          act[m * 64 + n] = (_Float16)fmaxf(acc[nt][r] + bn, 0.f);
        }
      }
    }

    // ---- layer 2: e2 = relu(e1 @ Wt2 + bt2)  (act -> act, f16) ----
    layer64(act, fbuf, wB, OFF_WT2, bt2G, lane, mrow, kbA, true);

    // ---- opacity trunk: o = relu(e2 @ Wo1 + bo1)  (act -> fbuf, f32) ----
    layer64(act, fbuf, wB, OFF_WO1, bo1G, lane, mrow, kbA, false);

    // opacity head: raw = o @ Wo2 + bo2 ; density = softplus(raw) ; alpha
    if (lane < 16) {
      float sacc = bo2G[0];
#pragma unroll 8
      for (int n = 0; n < HIDN; ++n) sacc += fbuf[lane * 64 + n] * Wo2G[n];
      float x = sacc;  // GAIN == 1
      float sp = (x > 20.f) ? x : log1pf(expf(x));
      alphaS[tt * 16 + lane] = 1.f - expf(-delta * sp);
    }

    // cin = e2 + ray_enc (in place, f16; same-wave LDS ordering)
#pragma unroll
    for (int it = 0; it < 32; ++it) {
      int e = lane + it * 32;
      act[e] = (_Float16)((float)act[e] + rayenc[e & 63]);
    }

    // ---- color trunk: h = relu(cin @ Wc1 + bc1)  (act -> fbuf, f32) ----
    layer64(act, fbuf, wB, OFF_WC1, bc1G, lane, mrow, kbA, false);

    // color head: col = sigmoid(h @ Wc2 + bc2)
    if (lane < 16) {
#pragma unroll
      for (int c = 0; c < 3; ++c) {
        float sacc = bc2G[c];
#pragma unroll 8
        for (int n = 0; n < HIDN; ++n) sacc += fbuf[lane * 64 + n] * Wc2G[n * 3 + c];
        colS[(tt * 16 + lane) * 3 + c] = 1.f / (1.f + expf(-sacc));
      }
    }
  }

  // ---- alpha compositing scan over the ray ----
  __syncthreads();
  if (tid == 0) {
    float T = 1.f, r0 = 0.f, r1 = 0.f, r2 = 0.f;
    for (int s2 = 0; s2 < SSN; ++s2) {
      float a = alphaS[s2];
      float w2 = T * a;
      r0 += w2 * colS[s2 * 3 + 0];
      r1 += w2 * colS[s2 * 3 + 1];
      r2 += w2 * colS[s2 * 3 + 2];
      T *= (1.f - a);
    }
    out[ray * 3 + 0] = r0;
    out[ray * 3 + 1] = r1;
    out[ray * 3 + 2] = r2;
    out[RR * 3 + ray] = 1.f - T;   // mask
  }
}

extern "C" void kernel_launch(void* const* d_in, const int* in_sizes, int n_in,
                              void* d_out, int out_size, void* d_ws, size_t ws_size,
                              hipStream_t stream) {
  (void)in_sizes; (void)n_in; (void)out_size;
  const float* org  = (const float*)d_in[0];
  const float* dirp = (const float*)d_in[1];
  const float* nearp= (const float*)d_in[2];
  const float* farp = (const float*)d_in[3];
  const float* grid = (const float*)d_in[4];
  const float* Wr   = (const float*)d_in[5];
  const float* br   = (const float*)d_in[6];
  const float* Wt1  = (const float*)d_in[7];
  const float* bt1  = (const float*)d_in[8];
  const float* Wt2  = (const float*)d_in[9];
  const float* bt2  = (const float*)d_in[10];
  const float* Wo1  = (const float*)d_in[11];
  const float* bo1  = (const float*)d_in[12];
  const float* Wo2  = (const float*)d_in[13];
  const float* bo2  = (const float*)d_in[14];
  const float* Wc1  = (const float*)d_in[15];
  const float* bc1  = (const float*)d_in[16];
  const float* Wc2  = (const float*)d_in[17];
  const float* bc2  = (const float*)d_in[18];
  float* outp = (float*)d_out;

  if (ws_size >= GRID_F16_BYTES) {
    // f16 grid in workspace: 128 MB working set => L2-resident gathers.
    _Float16* g16 = (_Float16*)d_ws;
    grid_to_f16<<<4096, 256, 0, stream>>>(grid, g16, GRID_ELEMS / 4);
    lightplane_render<_Float16><<<RR, 128, 0, stream>>>(org, dirp, nearp, farp,
        (const _Float16*)g16, Wr, br, Wt1, bt1, Wt2, bt2, Wo1, bo1, Wo2, bo2,
        Wc1, bc1, Wc2, bc2, outp);
  } else {
    lightplane_render<float><<<RR, 128, 0, stream>>>(org, dirp, nearp, farp, grid,
        Wr, br, Wt1, bt1, Wt2, bt2, Wo1, bo1, Wo2, bo2, Wc1, bc1, Wc2, bc2, outp);
  }
}
